// QKVAttention_77541339562393
// MI455X (gfx1250) — matrix operand order repaired and checked by reading, not compiled, not run on hardware
//
#include <hip/hip_runtime.h>
#include <float.h>
#include <stdint.h>

typedef __attribute__((ext_vector_type(16))) _Float16 v16h;
typedef __attribute__((ext_vector_type(8)))  _Float16 v8h;
typedef __attribute__((ext_vector_type(8)))  float    v8f;
typedef int v4i_vs __attribute__((vector_size(16)));   // matches builtin param type

#define N_CTX      1024
#define HEAD_DIM   64
#define N_HEADS    16
#define ROWS_PER_WAVE 16
#define WAVES      4
#define BLOCK_ROWS (ROWS_PER_WAVE * WAVES)   // 64 query rows per workgroup
#define KV_TILE    32
#define THREADS    (WAVES * 32)              // 128 threads = 4 wave32
#define KV_ITERS   (N_CTX / KV_TILE)         // 32

#if __has_builtin(__builtin_amdgcn_global_load_async_to_lds_b128) && \
    __has_builtin(__builtin_amdgcn_s_wait_asynccnt)
#define HAVE_ASYNC 1
#else
#define HAVE_ASYNC 0
#endif

__device__ __forceinline__ v16h join8(v8h lo, v8h hi) {
  return __builtin_shufflevector(lo, hi, 0,1,2,3,4,5,6,7,8,9,10,11,12,13,14,15);
}

__device__ __forceinline__ v8f wmma_f16(v16h a, v16h b, v8f c) {
  return __builtin_amdgcn_wmma_f32_16x16x32_f16(false, a, false, b, (short)0, c,
                                                false, false);
}

// --- VALU (DPP16) butterfly reductions across each 16-lane row -------------
// Masks {1,2,7,15} form an XOR basis of the 4-bit lane-in-row index, so this
// is a full 16-lane all-reduce that never crosses the half-wave boundary
// (lanes 0-15 / 16-31 hold different matrix rows in the WMMA C layout).
#define DPPF(x, ctrl)                                                         \
  __builtin_bit_cast(float, __builtin_amdgcn_update_dpp(                      \
      __builtin_bit_cast(int, (x)), __builtin_bit_cast(int, (x)),             \
      (ctrl), 0xf, 0xf, false))

__device__ __forceinline__ float row_max16(float x) {
  x = fmaxf(x, DPPF(x, 0xB1));   // quad_perm [1,0,3,2]  (xor 1)
  x = fmaxf(x, DPPF(x, 0x4E));   // quad_perm [2,3,0,1]  (xor 2)
  x = fmaxf(x, DPPF(x, 0x141));  // row_half_mirror      (xor 7)
  x = fmaxf(x, DPPF(x, 0x140));  // row_mirror           (xor 15)
  return x;
}
__device__ __forceinline__ float row_sum16(float x) {
  x += DPPF(x, 0xB1);
  x += DPPF(x, 0x4E);
  x += DPPF(x, 0x141);
  x += DPPF(x, 0x140);
  return x;
}

// --- CDNA5 async global->LDS (ASYNCcnt) with portable fallback -------------
__device__ __forceinline__ void async_cp16(const float* g, float* l) {
#if HAVE_ASYNC
  __builtin_amdgcn_global_load_async_to_lds_b128(
      (v4i_vs*)const_cast<float*>(g), (v4i_vs*)l, 0, 0);
#else
  *(float4*)l = *(const float4*)g;
#endif
}
__device__ __forceinline__ void async_wait_all() {
#if HAVE_ASYNC
  __builtin_amdgcn_s_wait_asynccnt(0);
#endif
}

__global__ __launch_bounds__(THREADS)
void qkv_attn_fa_kernel(const float* __restrict__ qkv,
                        const unsigned char* __restrict__ mask,
                        float* __restrict__ out)
{
  // Double-buffered K (fp16, transposed [j][d]) and V (raw fp32 [d][j], DMA'd)
  __shared__ __attribute__((aligned(32))) _Float16 ldsQ[WAVES][ROWS_PER_WAVE * HEAD_DIM];
  __shared__ __attribute__((aligned(32))) _Float16 ldsK[2][KV_TILE * HEAD_DIM];
  __shared__ __attribute__((aligned(16))) float    ldsV[2][HEAD_DIM * KV_TILE];
  __shared__ __attribute__((aligned(32))) _Float16 ldsP[WAVES][ROWS_PER_WAVE * KV_TILE];
  __shared__ __attribute__((aligned(16))) float    ldsO[HEAD_DIM * BLOCK_ROWS];

  const int tid  = threadIdx.x;
  const int lane = tid & 31;
  const int wave = tid >> 5;
  const int half = lane >> 4;
  const int l16  = lane & 15;

  const int bh = blockIdx.y;
  const int b  = bh >> 4;
  const int i0 = blockIdx.x * BLOCK_ROWS;

  const float scale = 0.3535533905932738f;  // 1/sqrt(sqrt(64))

  const float* qbase = qkv + (size_t)b * (3 * 1024) * N_CTX + (size_t)(bh & 15) * HEAD_DIM * N_CTX;
  const float* kbase = qbase + (size_t)1024 * N_CTX;
  const float* vbase = qbase + (size_t)2048 * N_CTX;

  // ---- Stage this wave's 16x64 Q tile as fp16 [i][d], pre-scaled ----
  {
    const int iw = i0 + wave * ROWS_PER_WAVE;
#pragma unroll
    for (int k = 0; k < 32; ++k) {
      int idx = k * 32 + lane;
      int dd = idx >> 4;
      int ii = idx & 15;
      ldsQ[wave][ii * HEAD_DIM + dd] = (_Float16)(qbase[(size_t)dd * N_CTX + iw + ii] * scale);
    }
  }

  // Persistent Q A-fragments (contiguous k = 16*half + s, matching B)
  const int M  = l16;
  const int kb = half * 16;
  const _Float16* qrow = &ldsQ[wave][M * HEAD_DIM];
  v16h Qf0 = join8(*(const v8h*)(qrow + kb),      *(const v8h*)(qrow + 8 + kb));
  v16h Qf1 = join8(*(const v8h*)(qrow + 32 + kb), *(const v8h*)(qrow + 40 + kb));

  // ---- Prologue: stage KV tile 0 into buffer 0 ----
#pragma unroll
  for (int c = 0; c < 4; ++c) {
    int chunk = c * THREADS + tid;      // 512 x 16B chunks per 64x32 fp32 tile
    int dd  = chunk >> 3;
    int off = (chunk & 7) * 4;
    async_cp16(vbase + (size_t)dd * N_CTX + off, &ldsV[0][dd * KV_TILE + off]);
    float4 kq = *(const float4*)(kbase + (size_t)dd * N_CTX + off);
    _Float16* kd = &ldsK[0][off * HEAD_DIM + dd];   // transpose [d][j] -> [j][d]
    kd[0]            = (_Float16)(kq.x * scale);
    kd[HEAD_DIM]     = (_Float16)(kq.y * scale);
    kd[2 * HEAD_DIM] = (_Float16)(kq.z * scale);
    kd[3 * HEAD_DIM] = (_Float16)(kq.w * scale);
  }
  async_wait_all();
  __syncthreads();

  v8f O0 = {}, O1 = {}, O2 = {}, O3 = {};
  float mrow[8], lrow[8];
#pragma unroll
  for (int r = 0; r < 8; ++r) { mrow[r] = -FLT_MAX; lrow[r] = 0.0f; }

  const unsigned char* mrow_mask = mask + (size_t)b * N_CTX;

  for (int it = 0; it < KV_ITERS; ++it) {
    const int cur = it & 1;
    const int j0  = it * KV_TILE;
    const bool more = (it + 1 < KV_ITERS);
    const _Float16* K_cur = ldsK[cur];
    const float*    V_cur = ldsV[cur];

    // ---- Issue next tile: V via async DMA to LDS, K into VGPRs ----
    float4 kr[4];
    if (more) {
      const int jn = j0 + KV_TILE;
#pragma unroll
      for (int c = 0; c < 4; ++c) {
        int chunk = c * THREADS + tid;
        int dd  = chunk >> 3;
        int off = (chunk & 7) * 4;
        async_cp16(vbase + (size_t)dd * N_CTX + jn + off,
                   &ldsV[cur ^ 1][dd * KV_TILE + off]);
        kr[c] = *(const float4*)(kbase + (size_t)dd * N_CTX + jn + off);
      }
    }

    // ---- S = Q^T K on WMMA (two 16x16 tiles, K-dim = head_dim in 2 chunks) ----
    const _Float16* kc0 = &K_cur[(l16)      * HEAD_DIM + half * 16];
    const _Float16* kc1 = &K_cur[(16 + l16) * HEAD_DIM + half * 16];
    v8f Sa = {}, Sb = {};
    Sa = wmma_f16(Qf0, *(const v16h*)(kc0),      Sa);
    Sa = wmma_f16(Qf1, *(const v16h*)(kc0 + 32), Sa);
    Sb = wmma_f16(Qf0, *(const v16h*)(kc1),      Sb);
    Sb = wmma_f16(Qf1, *(const v16h*)(kc1 + 32), Sb);

    const float bias_a = mrow_mask[j0 + l16]      ? 0.0f : -FLT_MAX;
    const float bias_b = mrow_mask[j0 + 16 + l16] ? 0.0f : -FLT_MAX;

    // ---- Online softmax in C layout; reductions are pure-VALU DPP ----
#pragma unroll
    for (int r = 0; r < 8; ++r) {
      float sa = Sa[r] + bias_a;
      float sb = Sb[r] + bias_b;
      float mn = fmaxf(mrow[r], row_max16(fmaxf(sa, sb)));
      float al = __expf(mrow[r] - mn);
      mrow[r] = mn;
      float ea = __expf(sa - mn);
      float eb = __expf(sb - mn);
      lrow[r] = lrow[r] * al + row_sum16(ea + eb);
      O0[r] *= al; O1[r] *= al; O2[r] *= al; O3[r] *= al;
      int i = r + half * 8;
      ldsP[wave][i * KV_TILE + l16]      = (_Float16)ea;
      ldsP[wave][i * KV_TILE + 16 + l16] = (_Float16)eb;
    }

    // ---- O += P * V^T : P as A-fragment, V B-fragments built with cvt ----
    const _Float16* prow = &ldsP[wave][M * KV_TILE];
    v16h Pf = join8(*(const v8h*)(prow + kb), *(const v8h*)(prow + 8 + kb));

    v16h Vf;
#define BUILD_VF(t)                                                           \
    {                                                                         \
      const float* vr = &V_cur[((t) * 16 + l16) * KV_TILE + half * 16];       \
      union { v16h v; _Float16 h[16]; } u_;                                   \
      _Pragma("unroll")                                                       \
      for (int e = 0; e < 16; ++e) u_.h[e] = (_Float16)vr[e];                 \
      Vf = u_.v;                                                              \
    }
    BUILD_VF(0); O0 = wmma_f16(Pf, Vf, O0);
    BUILD_VF(1); O1 = wmma_f16(Pf, Vf, O1);
    BUILD_VF(2); O2 = wmma_f16(Pf, Vf, O2);
    BUILD_VF(3); O3 = wmma_f16(Pf, Vf, O3);
#undef BUILD_VF

    // ---- Commit next K tile (scale+cvt+transpose) and close the pipeline ----
    if (more) {
#pragma unroll
      for (int c = 0; c < 4; ++c) {
        int chunk = c * THREADS + tid;
        int dd  = chunk >> 3;
        int off = (chunk & 7) * 4;
        _Float16* kd = &ldsK[cur ^ 1][off * HEAD_DIM + dd];
        kd[0]            = (_Float16)(kr[c].x * scale);
        kd[HEAD_DIM]     = (_Float16)(kr[c].y * scale);
        kd[2 * HEAD_DIM] = (_Float16)(kr[c].z * scale);
        kd[3 * HEAD_DIM] = (_Float16)(kr[c].w * scale);
      }
      async_wait_all();
    }
    __syncthreads();
  }

  // ---- Normalize; transpose through LDS for coalesced stores ----
#pragma unroll
  for (int r = 0; r < 8; ++r) {
    float inv = 1.0f / lrow[r];
    int iloc = wave * ROWS_PER_WAVE + r + half * 8;
    ldsO[(l16)      * BLOCK_ROWS + iloc] = O0[r] * inv;
    ldsO[(16 + l16) * BLOCK_ROWS + iloc] = O1[r] * inv;
    ldsO[(32 + l16) * BLOCK_ROWS + iloc] = O2[r] * inv;
    ldsO[(48 + l16) * BLOCK_ROWS + iloc] = O3[r] * inv;
  }
  __syncthreads();

#pragma unroll
  for (int k = 0; k < 32; ++k) {
    int idx = k * THREADS + tid;
    int dd = idx >> 6;
    int ii = idx & 63;
    out[(size_t)(bh * HEAD_DIM + dd) * N_CTX + i0 + ii] = ldsO[idx];
  }
}

extern "C" void kernel_launch(void* const* d_in, const int* in_sizes, int n_in,
                              void* d_out, int out_size, void* d_ws, size_t ws_size,
                              hipStream_t stream) {
  const float* qkv = (const float*)d_in[0];
  const unsigned char* mask = (const unsigned char*)d_in[1];
  float* out = (float*)d_out;

  int bsz = in_sizes[0] / (3 * 1024 * N_CTX);
  dim3 grid(N_CTX / BLOCK_ROWS, bsz * N_HEADS);
  qkv_attn_fa_kernel<<<grid, dim3(THREADS), 0, stream>>>(qkv, mask, out);
}
